// EGNNVectorField_61770219651342
// MI455X (gfx1250) — matrix operand order repaired and checked by reading, not compiled, not run on hardware
//
#include <hip/hip_runtime.h>
#include <hip/hip_bf16.h>
#include <math.h>

// ---------------- problem constants (from reference) ----------------
#define GRIDW 8
static const int Cc = 512;           // code / h dim
static const int Hh = 512;           // hidden dim
static const int Aa = 5;             // field heads
static const int Kk = 8;             // kNN
static const int Bb = 2;             // batch
static const int NPq = 4096;         // query points per batch
static const int NGg = GRIDW*GRIDW*GRIDW;   // 512 grid points
static const int Tt = NPq + NGg;     // 4608 nodes per batch
static const int Nn = Bb * Tt;       // 9216 nodes total
static const int Ee = Nn * Kk;       // 73728 edges

static const int W1_KP = 1040;       // eW1T padded K stride (1025 -> 1040)
static const int ET = 64;            // edges per block (edge kernel)
static const int NT = 64;            // nodes per block (node kernel)

typedef __bf16 bf16_t;
typedef __attribute__((ext_vector_type(16))) __bf16 v16bf;
typedef __attribute__((ext_vector_type(8)))  __bf16 v8bf;
typedef __attribute__((ext_vector_type(8)))  float  v8f;
typedef __attribute__((ext_vector_type(4)))  int    v4i;

__device__ __forceinline__ float silu_f(float v){
    return v / (1.0f + __expf(-v));
}

// ---------- gfx1250 async global->LDS copy path (probe-confirmed to exist) ----------
#if defined(__has_builtin)
# if __has_builtin(__builtin_amdgcn_global_load_async_to_lds_b128)
#  define HAVE_ASYNC_COPY 1
# endif
#endif

#ifdef HAVE_ASYNC_COPY
#define COPY16_G2L(dst, src)                                              \
    __builtin_amdgcn_global_load_async_to_lds_b128(                       \
        (__attribute__((address_space(1))) v4i*)(src),                    \
        (__attribute__((address_space(3))) v4i*)(dst), 0, 0)
__device__ __forceinline__ void wait_async(){
# if __has_builtin(__builtin_amdgcn_s_wait_asynccnt)
    __builtin_amdgcn_s_wait_asynccnt(0);
# else
    asm volatile("s_wait_asynccnt 0" ::: "memory");
# endif
}
#else
#define COPY16_G2L(dst, src) (*(v8bf*)(dst) = *(const v8bf*)(src))
__device__ __forceinline__ void wait_async(){}
#endif

// A-fragment (16x32 bf16): lane holds row M=lane%16, half=lane/16 selects K range.
// element e of v16bf: K = kb + half*16 + e
__device__ __forceinline__ v16bf load_afrag(const bf16_t* rowp, int kb, int half){
    union { v16bf v; v8bf h[2]; } u;
    u.h[0] = *(const v8bf*)(rowp + kb + half*16);
    u.h[1] = *(const v8bf*)(rowp + kb + half*16 + 8);
    return u.v;
}

// B-fragment (32x16 bf16): lane holds col N=lane%16; lanes 0-15 K=kb..kb+15,
// lanes 16-31 K=kb+16..kb+31. Weights stored transposed [N][K] -> one 32B read.
__device__ __forceinline__ v16bf load_bfrag(const bf16_t* WT, int strideT, int n, int kb, int half){
    return *(const v16bf*)(WT + (size_t)n*strideT + kb + half*16);
}

__device__ __forceinline__ v8f wmma_bf16(v16bf a, v16bf b, v8f c){
    return __builtin_amdgcn_wmma_f32_16x16x32_bf16(
        false, a, false, b, (short)0, c, false, false);
}

// ---------------- init kernels ----------------
__global__ void init_nodes_kernel(const float* __restrict__ qp, const float* __restrict__ gp,
                                  float* __restrict__ x, float* __restrict__ nf){
    int n = blockIdx.x*256 + threadIdx.x;
    if (n >= Nn) return;
    int b = n / Tt, w = n % Tt;
    float vx, vy, vz, f;
    if (w < NPq){
        const float* p = qp + ((size_t)b*NPq + w)*3;
        vx=p[0]; vy=p[1]; vz=p[2]; f = 1.0f;
    } else {
        const float* p = gp + (size_t)(w - NPq)*3;
        vx=p[0]; vy=p[1]; vz=p[2]; f = 0.0f;
    }
    x[n*3+0]=vx; x[n*3+1]=vy; x[n*3+2]=vz; nf[n]=f;
}

__global__ void init_h_kernel(const float* __restrict__ codes,
                              float* __restrict__ h32, bf16_t* __restrict__ hbf){
    int idx = blockIdx.x*256 + threadIdx.x;
    if (idx >= Nn*Cc) return;
    int n = idx >> 9, c = idx & 511;
    int b = n / Tt, w = n % Tt;
    float v = (w < NPq) ? 0.0f : codes[((size_t)b*NGg + (w - NPq))*Cc + c];
    h32[idx] = v;
    hbf[idx] = (bf16_t)v;
}

// transpose+convert weights: W[K][Nw] f32 -> WT[Nw][strideT] bf16
__global__ void transpose_bf16_kernel(const float* __restrict__ W, bf16_t* __restrict__ WT,
                                      int K, int Nw, int strideT){
    int idx = blockIdx.x*256 + threadIdx.x;
    if (idx >= K*Nw) return;
    int k = idx / Nw, nn = idx % Nw;
    WT[(size_t)nn*strideT + k] = (bf16_t)W[idx];
}

// ---------------- edge MLP (WMMA) ----------------
// 64-edge tile per workgroup, 256 threads = 8 waves.
// Each wave: 4 N-tiles, and per N-tile 4 M-subtiles reusing each B-fragment 4x.
template<int OUTX>
__global__ __launch_bounds__(256)
void edge_mlp_kernel(const float* __restrict__ x, const bf16_t* __restrict__ hbf,
                     const int* __restrict__ rowidx,
                     const bf16_t* __restrict__ W1T, const float* __restrict__ b1,
                     const bf16_t* __restrict__ W2T, const float* __restrict__ b2,
                     const float* __restrict__ cW,  const float* __restrict__ cb,
                     bf16_t* __restrict__ m_out, float* __restrict__ coef_out){
    __shared__ __align__(16) bf16_t feat[ET][W1_KP];   // gathered [h_row|h_col]; reused as m2 in [i][0..511]
    __shared__ __align__(16) bf16_t m1[ET][Hh];
    __shared__ float sdist[ET];
    __shared__ int   srow[ET];
    __shared__ float spart[ET][OUTX][8];

    const int tid = threadIdx.x;
    const int e0  = blockIdx.x * ET;

    if (tid < ET){
        int e = e0 + tid;
        int r = rowidx[e];
        int cn = e >> 3;                    // col[e] == e / K (structured edge list)
        srow[tid] = r;
        float dx = x[r*3+0]-x[cn*3+0];
        float dy = x[r*3+1]-x[cn*3+1];
        float dz = x[r*3+2]-x[cn*3+2];
        sdist[tid] = sqrtf(dx*dx + dy*dy + dz*dz);
    }
    __syncthreads();

    // stage [h[row] | h[col]] -> LDS (16B chunks, async)
    for (int idx = tid; idx < ET*64; idx += 256){
        int i  = idx >> 6;
        int c8 = (idx & 63) * 8;
        int e  = e0 + i;
        size_t r  = (size_t)srow[i];
        size_t cn = (size_t)(e >> 3);
        COPY16_G2L(&feat[i][c8],      &hbf[r *Cc + c8]);
        COPY16_G2L(&feat[i][Cc + c8], &hbf[cn*Cc + c8]);
    }
    wait_async();
    __syncthreads();

    const int lane = tid & 31, w = tid >> 5;
    const int half = lane >> 4, nl = lane & 15;

    // GEMM1: [64 x 1024] x [1024 x 512] + dist rank-1 + bias, SiLU -> m1
    for (int j = 0; j < 4; ++j){
        int n = (w*4 + j)*16 + nl;
        v8f acc0 = {}, acc1 = {}, acc2 = {}, acc3 = {};
        for (int kb = 0; kb < 2*Cc; kb += 32){
            v16bf bfr = load_bfrag(W1T, W1_KP, n, kb, half);
            acc0 = wmma_bf16(load_afrag(&feat[ 0 + nl][0], kb, half), bfr, acc0);
            acc1 = wmma_bf16(load_afrag(&feat[16 + nl][0], kb, half), bfr, acc1);
            acc2 = wmma_bf16(load_afrag(&feat[32 + nl][0], kb, half), bfr, acc2);
            acc3 = wmma_bf16(load_afrag(&feat[48 + nl][0], kb, half), bfr, acc3);
        }
        float wlast = (float)W1T[(size_t)n*W1_KP + 2*Cc];   // dist row of eW1
        float bias  = b1[n];
        #pragma unroll
        for (int r = 0; r < 8; ++r){
            int M = r + half*8;
            m1[ 0 + M][n] = (bf16_t)silu_f(acc0[r] + sdist[ 0 + M]*wlast + bias);
            m1[16 + M][n] = (bf16_t)silu_f(acc1[r] + sdist[16 + M]*wlast + bias);
            m1[32 + M][n] = (bf16_t)silu_f(acc2[r] + sdist[32 + M]*wlast + bias);
            m1[48 + M][n] = (bf16_t)silu_f(acc3[r] + sdist[48 + M]*wlast + bias);
        }
    }
    __syncthreads();

    // GEMM2: [64 x 512] x [512 x 512], SiLU -> m2 (reuse feat buffer)
    for (int j = 0; j < 4; ++j){
        int n = (w*4 + j)*16 + nl;
        v8f acc0 = {}, acc1 = {}, acc2 = {}, acc3 = {};
        for (int kb = 0; kb < Hh; kb += 32){
            v16bf bfr = load_bfrag(W2T, Hh, n, kb, half);
            acc0 = wmma_bf16(load_afrag(&m1[ 0 + nl][0], kb, half), bfr, acc0);
            acc1 = wmma_bf16(load_afrag(&m1[16 + nl][0], kb, half), bfr, acc1);
            acc2 = wmma_bf16(load_afrag(&m1[32 + nl][0], kb, half), bfr, acc2);
            acc3 = wmma_bf16(load_afrag(&m1[48 + nl][0], kb, half), bfr, acc3);
        }
        float bias = b2[n];
        #pragma unroll
        for (int r = 0; r < 8; ++r){
            int M = r + half*8;
            feat[ 0 + M][n] = (bf16_t)silu_f(acc0[r] + bias);
            feat[16 + M][n] = (bf16_t)silu_f(acc1[r] + bias);
            feat[32 + M][n] = (bf16_t)silu_f(acc2[r] + bias);
            feat[48 + M][n] = (bf16_t)silu_f(acc3[r] + bias);
        }
    }
    __syncthreads();

    // store m2 tile to global (contiguous 16B stores)
    for (int idx = tid; idx < ET*64; idx += 256){
        int i  = idx >> 6;
        int c8 = (idx & 63) * 8;
        *(v8bf*)&m_out[((size_t)(e0 + i))*Hh + c8] = *(const v8bf*)&feat[i][c8];
    }
    // GEMM3 (tiny): coef = m2 @ cW + cb, deterministic 8-way partial tree
    for (int idx = tid; idx < ET*OUTX*8; idx += 256){
        int seg = idx & 7;
        int t   = idx >> 3;
        int a   = t % OUTX;
        int i   = t / OUTX;
        int k0  = seg * 64;
        float s = 0.f;
        for (int k = 0; k < 64; ++k)
            s += (float)feat[i][k0 + k] * cW[(k0 + k)*OUTX + a];
        spart[i][a][seg] = s;
    }
    __syncthreads();
    for (int idx = tid; idx < ET*OUTX; idx += 256){
        int a = idx % OUTX, i = idx / OUTX;
        float s = cb[a];
        #pragma unroll
        for (int g = 0; g < 8; ++g) s += spart[i][a][g];
        coef_out[(size_t)(e0 + i)*OUTX + a] = s;
    }
}

// ---------------- aggregation (K=8 consecutive edges per node) ----------------
__global__ void aggr_m_kernel(const bf16_t* __restrict__ m, bf16_t* __restrict__ maggr){
    int idx = blockIdx.x*256 + threadIdx.x;
    if (idx >= Nn*Hh) return;
    int n = idx >> 9, c = idx & 511;
    float s = 0.f;
    #pragma unroll
    for (int j = 0; j < Kk; ++j)
        s += (float)m[((size_t)(n*Kk + j))*Hh + c];
    maggr[idx] = (bf16_t)(s * 0.125f);
}

__global__ void dx_kernel(const float* __restrict__ x, const int* __restrict__ rowidx,
                          const float* __restrict__ coef, float* __restrict__ dxbuf){
    int n = blockIdx.x*256 + threadIdx.x;
    if (n >= Nn) return;
    float ax = x[n*3], ay = x[n*3+1], az = x[n*3+2];
    float sx = 0.f, sy = 0.f, sz = 0.f;
    #pragma unroll
    for (int j = 0; j < Kk; ++j){
        int e = n*Kk + j;
        int r = rowidx[e];
        float rx = x[r*3]-ax, ry = x[r*3+1]-ay, rz = x[r*3+2]-az;
        float d  = sqrtf(rx*rx + ry*ry + rz*rz);
        float inv = 1.0f / (d + 1e-8f);
        float c = coef[e];
        sx += c*rx*inv; sy += c*ry*inv; sz += c*rz*inv;
    }
    dxbuf[n*3+0] = sx*0.125f; dxbuf[n*3+1] = sy*0.125f; dxbuf[n*3+2] = sz*0.125f;
}

__global__ void x_update_kernel(float* __restrict__ x, const float* __restrict__ dxbuf,
                                const float* __restrict__ nf){
    int idx = blockIdx.x*256 + threadIdx.x;
    if (idx >= Nn*3) return;
    x[idx] += dxbuf[idx] * nf[idx/3];
}

// ---------------- node MLP (WMMA), 64-node tiles ----------------
__global__ __launch_bounds__(256)
void node_mlp_kernel(const bf16_t* __restrict__ hbf, const bf16_t* __restrict__ maggr,
                     const bf16_t* __restrict__ W1T, const float* __restrict__ b1,
                     const bf16_t* __restrict__ W2T, const float* __restrict__ b2,
                     float* __restrict__ h32, bf16_t* __restrict__ hbf_out,
                     const float* __restrict__ nf){
    __shared__ __align__(16) bf16_t feat[NT][2*Cc];
    __shared__ __align__(16) bf16_t m1[NT][Hh];
    const int tid = threadIdx.x;
    const int nb0 = blockIdx.x * NT;

    for (int idx = tid; idx < NT*64; idx += 256){
        int i  = idx >> 6;
        int c8 = (idx & 63) * 8;
        size_t node = (size_t)(nb0 + i);
        COPY16_G2L(&feat[i][c8],      &hbf  [node*Cc + c8]);
        COPY16_G2L(&feat[i][Cc + c8], &maggr[node*Hh + c8]);
    }
    wait_async();
    __syncthreads();

    const int lane = tid & 31, w = tid >> 5;
    const int half = lane >> 4, nl = lane & 15;

    for (int j = 0; j < 4; ++j){
        int n = (w*4 + j)*16 + nl;
        v8f acc0 = {}, acc1 = {}, acc2 = {}, acc3 = {};
        for (int kb = 0; kb < 2*Cc; kb += 32){
            v16bf bfr = load_bfrag(W1T, 2*Cc, n, kb, half);
            acc0 = wmma_bf16(load_afrag(&feat[ 0 + nl][0], kb, half), bfr, acc0);
            acc1 = wmma_bf16(load_afrag(&feat[16 + nl][0], kb, half), bfr, acc1);
            acc2 = wmma_bf16(load_afrag(&feat[32 + nl][0], kb, half), bfr, acc2);
            acc3 = wmma_bf16(load_afrag(&feat[48 + nl][0], kb, half), bfr, acc3);
        }
        float bias = b1[n];
        #pragma unroll
        for (int r = 0; r < 8; ++r){
            int M = r + half*8;
            m1[ 0 + M][n] = (bf16_t)silu_f(acc0[r] + bias);
            m1[16 + M][n] = (bf16_t)silu_f(acc1[r] + bias);
            m1[32 + M][n] = (bf16_t)silu_f(acc2[r] + bias);
            m1[48 + M][n] = (bf16_t)silu_f(acc3[r] + bias);
        }
    }
    __syncthreads();

    for (int j = 0; j < 4; ++j){
        int n = (w*4 + j)*16 + nl;
        v8f acc0 = {}, acc1 = {}, acc2 = {}, acc3 = {};
        for (int kb = 0; kb < Hh; kb += 32){
            v16bf bfr = load_bfrag(W2T, Hh, n, kb, half);
            acc0 = wmma_bf16(load_afrag(&m1[ 0 + nl][0], kb, half), bfr, acc0);
            acc1 = wmma_bf16(load_afrag(&m1[16 + nl][0], kb, half), bfr, acc1);
            acc2 = wmma_bf16(load_afrag(&m1[32 + nl][0], kb, half), bfr, acc2);
            acc3 = wmma_bf16(load_afrag(&m1[48 + nl][0], kb, half), bfr, acc3);
        }
        float bias = b2[n];
        #pragma unroll
        for (int r = 0; r < 8; ++r){
            int M = r + half*8;
            #pragma unroll
            for (int ms = 0; ms < 4; ++ms){
                float av = (ms==0?acc0[r]:ms==1?acc1[r]:ms==2?acc2[r]:acc3[r]);
                int node = nb0 + ms*16 + M;
                float f = nf[node];
                size_t offn = (size_t)node*Cc + n;
                float hv = h32[offn] + (av + bias) * f;
                h32[offn] = hv;
                hbf_out[offn] = (bf16_t)hv;
            }
        }
    }
}

// ---------------- field output ----------------
__global__ void field_out_kernel(const float* __restrict__ x, const int* __restrict__ rowidx,
                                 const float* __restrict__ coef5, const float* __restrict__ qp,
                                 float* __restrict__ out){
    int idx = blockIdx.x*256 + threadIdx.x;
    if (idx >= Bb*NPq*Aa) return;
    int a = idx % Aa;
    int t = idx / Aa;
    int b = t / NPq, i = t % NPq;
    int n = b*Tt + i;
    float ax = x[n*3], ay = x[n*3+1], az = x[n*3+2];
    float sx = 0.f, sy = 0.f, sz = 0.f;
    #pragma unroll
    for (int j = 0; j < Kk; ++j){
        int e = n*Kk + j;
        int r = rowidx[e];
        float rx = x[r*3]-ax, ry = x[r*3+1]-ay, rz = x[r*3+2]-az;
        float d  = sqrtf(rx*rx + ry*ry + rz*rz);
        float inv = 1.0f / (d + 1e-8f);
        float c = coef5[(size_t)e*Aa + a];
        sx += c*rx*inv; sy += c*ry*inv; sz += c*rz*inv;
    }
    const float* q = qp + ((size_t)b*NPq + i)*3;
    float* o = out + (size_t)idx*3;
    o[0] = ax + sx*0.125f - q[0];
    o[1] = ay + sy*0.125f - q[1];
    o[2] = az + sz*0.125f - q[2];
}

// ---------------- host side ----------------
extern "C" void kernel_launch(void* const* d_in, const int* in_sizes, int n_in,
                              void* d_out, int out_size, void* d_ws, size_t ws_size,
                              hipStream_t stream){
    const float* qp     = (const float*)d_in[0];
    const float* codes  = (const float*)d_in[1];
    const float* gp     = (const float*)d_in[2];
    const int*   rowidx = (const int*)  d_in[3];   // edge_index[0] (first E ints)

    auto P = [&](int layer, int item)->const float*{
        return (const float*)d_in[4 + layer*10 + item];
    };

    char* ws = (char*)d_ws;
    size_t off = 0;
    auto alloc = [&](size_t bytes)->void*{
        off = (off + 255) & ~(size_t)255;
        void* p = ws + off;
        off += bytes;
        return p;
    };

    float*  x     = (float*) alloc((size_t)Nn*3*sizeof(float));
    float*  nf    = (float*) alloc((size_t)Nn*sizeof(float));
    float*  h32   = (float*) alloc((size_t)Nn*Cc*sizeof(float));
    bf16_t* hbf   = (bf16_t*)alloc((size_t)Nn*Cc*sizeof(bf16_t));
    bf16_t* mbuf  = (bf16_t*)alloc((size_t)Ee*Hh*sizeof(bf16_t));
    bf16_t* maggr = (bf16_t*)alloc((size_t)Nn*Hh*sizeof(bf16_t));
    float*  coef  = (float*) alloc((size_t)Ee*Aa*sizeof(float));
    float*  dxbuf = (float*) alloc((size_t)Nn*3*sizeof(float));

    bf16_t* eW1T[4]; bf16_t* eW2T[4]; bf16_t* nW1T[4]; bf16_t* nW2T[4];
    for (int l = 0; l < 4; ++l){
        eW1T[l] = (bf16_t*)alloc((size_t)Hh*W1_KP*sizeof(bf16_t));
        eW2T[l] = (bf16_t*)alloc((size_t)Hh*Hh*sizeof(bf16_t));
        nW1T[l] = (bf16_t*)alloc((size_t)Hh*(2*Cc)*sizeof(bf16_t));
        nW2T[l] = (bf16_t*)alloc((size_t)Cc*Hh*sizeof(bf16_t));
    }

    init_nodes_kernel<<<(Nn+255)/256, 256, 0, stream>>>(qp, gp, x, nf);
    init_h_kernel<<<((size_t)Nn*Cc+255)/256, 256, 0, stream>>>(codes, h32, hbf);

    for (int l = 0; l < 4; ++l){
        transpose_bf16_kernel<<<((2*Cc+1)*Hh+255)/256, 256, 0, stream>>>(P(l,0), eW1T[l], 2*Cc+1, Hh, W1_KP);
        transpose_bf16_kernel<<<(Hh*Hh+255)/256,       256, 0, stream>>>(P(l,2), eW2T[l], Hh,     Hh, Hh);
        transpose_bf16_kernel<<<(2*Cc*Hh+255)/256,     256, 0, stream>>>(P(l,6), nW1T[l], 2*Cc,   Hh, 2*Cc);
        transpose_bf16_kernel<<<(Hh*Cc+255)/256,       256, 0, stream>>>(P(l,8), nW2T[l], Hh,     Cc, Hh);
    }

    for (int l = 0; l < 3; ++l){
        edge_mlp_kernel<1><<<Ee/ET, 256, 0, stream>>>(
            x, hbf, rowidx,
            eW1T[l], P(l,1), eW2T[l], P(l,3), P(l,4), P(l,5),
            mbuf, coef);
        aggr_m_kernel<<<((size_t)Nn*Hh+255)/256, 256, 0, stream>>>(mbuf, maggr);
        dx_kernel<<<(Nn+255)/256, 256, 0, stream>>>(x, rowidx, coef, dxbuf);
        x_update_kernel<<<(Nn*3+255)/256, 256, 0, stream>>>(x, dxbuf, nf);
        node_mlp_kernel<<<Nn/NT, 256, 0, stream>>>(
            hbf, maggr, nW1T[l], P(l,7), nW2T[l], P(l,9), h32, hbf, nf);
    }

    edge_mlp_kernel<5><<<Ee/ET, 256, 0, stream>>>(
        x, hbf, rowidx,
        eW1T[3], P(3,1), eW2T[3], P(3,3), P(3,4), P(3,5),
        mbuf, coef);
    field_out_kernel<<<(Bb*NPq*Aa+255)/256, 256, 0, stream>>>(x, rowidx, coef, qp, (float*)d_out);
}
